// ConcatAttention_2482491097108
// MI455X (gfx1250) — matrix operand order repaired and checked by reading, not compiled, not run on hardware
//
#include <hip/hip_runtime.h>
#include <math.h>

typedef __attribute__((ext_vector_type(16))) __bf16 bf16x16;
typedef __attribute__((ext_vector_type(8)))  __bf16 bf16x8;
typedef __attribute__((ext_vector_type(8)))  float  f32x8;

#define Lq 4096
#define Bq 32
#define Hq 512
#define MT 4            // 4 x 16-row WMMA tiles per workgroup -> 64 M rows / WG
#define APAD 48         // padded LDS row stride in bf16 (96 B: 32-B aligned frags, bank spread)

// ---------------- prologue kernels ----------------

// WkT[n][k] = bf16( W[H + k][n] )   (transpose + downconvert, 512x512, L2-resident)
__global__ void wkt_kernel(const float* __restrict__ W, __bf16* __restrict__ wkT) {
    int idx = blockIdx.x * blockDim.x + threadIdx.x;        // 512*512 threads
    int n = idx >> 9, k = idx & 511;
    wkT[((size_t)n << 9) + k] = (__bf16)W[((size_t)(Hq + k) << 9) + n];
}

// q_proj[b][h] = query[b][:] . Wq[:][h]   (32x512, tiny)
__global__ void qproj_kernel(const float* __restrict__ query, const float* __restrict__ W,
                             float* __restrict__ qproj) {
    int idx = blockIdx.x * blockDim.x + threadIdx.x;        // 32*512 threads
    int b = idx >> 9, h = idx & 511;
    float s = 0.f;
    for (int i = 0; i < Hq; ++i)
        s = fmaf(query[(b << 9) + i], W[((size_t)i << 9) + h], s);
    qproj[idx] = s;
}

// ---------------- fused main kernel ----------------
// scores[b][l] = sum_h v[h] * tanh(qproj[b][h] + keyval_row[(l*B+b)] . WkT[h][:])
__launch_bounds__(256)
__global__ void fused_kproj_score(const float* __restrict__ keyval,   // (L*B, 512) f32
                                  const __bf16* __restrict__ wkT,     // (512 n, 512 k) bf16
                                  const float* __restrict__ qproj,    // (32, 512) f32
                                  const float* __restrict__ vvec,     // (512,) f32
                                  float* __restrict__ scores)         // (32, 4096) raw
{
    // A staged in LDS as bf16, in natural k order (contiguous k = 16*hi + s):
    // per row m: [K0..7, K8..15, K16..23, K24..31]  -> lane(hi) reads 32 B at offset hi*16 elems
    __shared__ __bf16 abuf[2][64 * APAD];   // 2 x 6 KB
    __shared__ float  vsh[Hq];
    __shared__ float  rowsum[MT * 16];

    const int tid = threadIdx.x;
    for (int i = tid; i < Hq; i += 256) vsh[i] = vvec[i];
    if (tid < MT * 16) rowsum[tid] = 0.f;

    const int lane = tid & 31;
    const int wave = tid >> 5;          // 8 waves: wave owns N columns [wave*64, wave*64+64)
    const int lm   = lane & 15;
    const int hi   = lane >> 4;         // 0: lanes 0-15, 1: lanes 16-31
    const long mbase = (long)blockIdx.x * (MT * 16);

    // --- staging roles: thread t handles (row = t>>2, K-group q = t&3) of each 64x32 chunk ---
    const int srow = tid >> 2;
    const int q    = tid & 3;                                   // K-group {0..7,8..15,16..23,24..31}
    const int dstoff = q << 3;                                  // 0->0, 1->8, 2->16, 3->24
    const float* gsrc = keyval + (size_t)(mbase + srow) * Hq + q * 8;
    __bf16* const sdst0 = &abuf[0][srow * APAD + dstoff];
    __bf16* const sdst1 = &abuf[1][srow * APAD + dstoff];

    f32x8 acc[MT][4];
    #pragma unroll
    for (int mi = 0; mi < MT; ++mi)
        #pragma unroll
        for (int t = 0; t < 4; ++t)
            acc[mi][t] = (f32x8)0.f;

    // per-lane base into WkT: column n = wave*64 + t*16 + lm, k-offset hi*16
    const __bf16* bcol = wkT + (((size_t)(wave << 6) + lm) << 9) + (hi << 4);

    // ---- prime chunk 0 ----
    {
        const f32x8 d = *(const f32x8*)(gsrc);
        bf16x8 o;
        #pragma unroll
        for (int i = 0; i < 8; ++i) o[i] = (__bf16)d[i];
        *(bf16x8*)sdst0 = o;
    }
    __syncthreads();

    for (int it = 0; it < 16; ++it) {
        const int kb = it * 32;

        // issue next chunk's global loads early (overlaps WMMA)
        f32x8 d;
        if (it + 1 < 16) d = *(const f32x8*)(gsrc + (it + 1) * 32);
        if (it + 2 < 16) __builtin_prefetch(gsrc + (it + 2) * 32, 0, 2);

        // A fragments straight out of LDS (fragment-ordered, 32-B aligned, no packing VALU)
        const __bf16* const cur = &abuf[it & 1][0];
        bf16x16 am[MT];
        #pragma unroll
        for (int mi = 0; mi < MT; ++mi)
            am[mi] = *(const bf16x16*)(cur + (mi * 16 + lm) * APAD + (hi << 4));

        // B fragments: 32x16 bf16 — lane holds contiguous 16-k column chunk of WkT row n
        #pragma unroll
        for (int t = 0; t < 4; ++t) {
            const bf16x16 bf = *(const bf16x16*)(bcol + ((size_t)(t * 16) << 9) + kb);
            #pragma unroll
            for (int mi = 0; mi < MT; ++mi)
                acc[mi][t] = __builtin_amdgcn_wmma_f32_16x16x32_bf16(
                    false, am[mi], false, bf, (short)0, acc[mi][t], false, false);
        }

        // convert + store next chunk into the other buffer
        if (it + 1 < 16) {
            bf16x8 o;
            #pragma unroll
            for (int i = 0; i < 8; ++i) o[i] = (__bf16)d[i];
            *(bf16x8*)((it & 1) ? sdst0 : sdst1) = o;
        }
        __syncthreads();
    }

    // Epilogue: C layout — VGPR r: M = r + hi*8, N = lm (+ tile offset)
    #pragma unroll
    for (int mi = 0; mi < MT; ++mi) {
        #pragma unroll
        for (int r = 0; r < 8; ++r) {
            const int m = mi * 16 + r + hi * 8;
            const int b = (int)((mbase + m) & (Bq - 1));
            float s = 0.f;
            #pragma unroll
            for (int t = 0; t < 4; ++t) {
                const int n = (wave << 6) + t * 16 + lm;
                s += vsh[n] * tanhf(qproj[(b << 9) + n] + acc[mi][t][r]);
            }
            // reduce the 16 N-lanes inside each half-wave (masks < 16 keep halves separate)
            s += __shfl_xor(s, 1, 32);
            s += __shfl_xor(s, 2, 32);
            s += __shfl_xor(s, 4, 32);
            s += __shfl_xor(s, 8, 32);
            if (lm == 0) atomicAdd(&rowsum[m], s);   // ds_add_f32, across 8 waves
        }
    }
    __syncthreads();

    if (tid < MT * 16) {
        const long g = mbase + tid;
        const int  b = (int)(g & (Bq - 1));
        const long l = g >> 5;
        scores[(size_t)b * Lq + l] = rowsum[tid];
    }
}

// ---------------- in-place row softmax over L ----------------
__global__ void softmax_rows(float* __restrict__ out) {
    __shared__ float red[256];
    const int b = blockIdx.x;
    float* row = out + (size_t)b * Lq;
    const int tid = threadIdx.x;

    float m = -INFINITY;
    for (int i = tid; i < Lq; i += 256) m = fmaxf(m, row[i]);
    red[tid] = m; __syncthreads();
    for (int s = 128; s > 0; s >>= 1) {
        if (tid < s) red[tid] = fmaxf(red[tid], red[tid + s]);
        __syncthreads();
    }
    const float mx = red[0];
    __syncthreads();

    float sum = 0.f;
    for (int i = tid; i < Lq; i += 256) sum += __expf(row[i] - mx);
    red[tid] = sum; __syncthreads();
    for (int s = 128; s > 0; s >>= 1) {
        if (tid < s) red[tid] += red[tid + s];
        __syncthreads();
    }
    const float inv = 1.0f / red[0];
    __syncthreads();

    for (int i = tid; i < Lq; i += 256) row[i] = __expf(row[i] - mx) * inv;
}

// ---------------- launcher ----------------
extern "C" void kernel_launch(void* const* d_in, const int* in_sizes, int n_in,
                              void* d_out, int out_size, void* d_ws, size_t ws_size,
                              hipStream_t stream) {
    const float* query  = (const float*)d_in[0];   // (32, 512)
    const float* keyval = (const float*)d_in[1];   // (4096, 32, 512)
    const float* W      = (const float*)d_in[2];   // (1024, 512)
    const float* vvec   = (const float*)d_in[3];   // (512,)
    float* out = (float*)d_out;                    // (32, 1, 4096)

    __bf16* wkT   = (__bf16*)d_ws;                               // 512*512*2 = 512 KB
    float*  qproj = (float*)((char*)d_ws + (size_t)Hq * Hq * 2); // 32*512*4  = 64 KB

    wkt_kernel  <<<(Hq * Hq) / 256, 256, 0, stream>>>(W, wkT);
    qproj_kernel<<<(Bq * Hq) / 256, 256, 0, stream>>>(query, W, qproj);

    const int nWG = (Lq * Bq) / (MT * 16);   // 2048 workgroups
    fused_kproj_score<<<nWG, 256, 0, stream>>>(keyval, wkT, qproj, vvec, out);

    softmax_rows<<<Bq, 256, 0, stream>>>(out);
}
